// SparseMoE_55207509622871
// MI455X (gfx1250) — matrix operand order repaired and checked by reading, not compiled, not run on hardware
//
#include <hip/hip_runtime.h>

// ---------------------------------------------------------------------------
// Problem constants (match reference)
// ---------------------------------------------------------------------------
#define BB 16
#define TT 2048
#define CC 384
#define EE 8
#define DFF 1536
#define NN (BB * TT)          // 32768 tokens
#define CAP 8192              // N*TOP_K/E*CAP_F
#define CHUNK 1024
#define NCHUNK (NN / CHUNK)   // 32

// FFN tiling
#define MT 64                 // rows per block
#define XP 392                // x_tile LDS pitch (ushorts), 784B rows (16B aligned, bank-spread)
#define HP 40                 // H / w2s LDS pitch (ushorts), 80B rows
#define W1P 392               // w1 slab pitch

typedef __attribute__((ext_vector_type(16))) __bf16 v16bf;
typedef __attribute__((ext_vector_type(8)))  float  v8f;
typedef int v4i __attribute__((vector_size(16)));

union V16U { v16bf v; uint4 q[2]; };

#define GLOBAL_AS __attribute__((address_space(1)))
#define LDS_AS    __attribute__((address_space(3)))

__device__ __forceinline__ unsigned short f2bf(float f) {
  unsigned int u = __float_as_uint(f);
  u += 0x7FFFu + ((u >> 16) & 1u);   // round-to-nearest-even
  return (unsigned short)(u >> 16);
}

// 16B async copy global -> LDS (ASYNCcnt-tracked)
__device__ __forceinline__ void async_cp16(const unsigned short* g, unsigned short* l) {
#if __has_builtin(__builtin_amdgcn_global_load_async_to_lds_b128)
  __builtin_amdgcn_global_load_async_to_lds_b128(
      (GLOBAL_AS v4i*)g, (LDS_AS v4i*)l, 0, 0);
#else
  unsigned lo = (unsigned)(unsigned long long)l;          // LDS byte offset (low 32 bits)
  unsigned long long ga = (unsigned long long)g;
  asm volatile("global_load_async_to_lds_b128 %0, %1, off"
               :: "v"(lo), "v"(ga) : "memory");
#endif
}

__device__ __forceinline__ void wait_async0() {
#if __has_builtin(__builtin_amdgcn_s_wait_asynccnt)
  __builtin_amdgcn_s_wait_asynccnt(0);
#else
  asm volatile("s_wait_asynccnt 0x0" ::: "memory");
#endif
}
__device__ __forceinline__ void wait_async6() {
#if __has_builtin(__builtin_amdgcn_s_wait_asynccnt)
  __builtin_amdgcn_s_wait_asynccnt(6);
#else
  asm volatile("s_wait_asynccnt 0x6" ::: "memory");
#endif
}

// ---------------------------------------------------------------------------
// Kernel 0: zero the output
// ---------------------------------------------------------------------------
__global__ __launch_bounds__(256) void k_zero(float* __restrict__ p, int n4) {
  int i = blockIdx.x * 256 + threadIdx.x;
  if (i < n4) {
    float4 z = make_float4(0.f, 0.f, 0.f, 0.f);
    ((float4*)p)[i] = z;
  }
}

// ---------------------------------------------------------------------------
// Kernel 1: routing. One wave per token; gate = sigmoid(v1 - v2).
// ---------------------------------------------------------------------------
__global__ __launch_bounds__(256) void k_route(
    const float* __restrict__ x, const float* __restrict__ noise,
    const float* __restrict__ w_route, const float* __restrict__ b_route,
    const float* __restrict__ w_noise, const float* __restrict__ b_noise,
    int* __restrict__ top1, float* __restrict__ gate) {
  int wave = threadIdx.x >> 5;
  int lane = threadIdx.x & 31;
  int n = blockIdx.x * 8 + wave;
  if (n >= NN) return;

  float ar[EE], an[EE];
#pragma unroll
  for (int e = 0; e < EE; ++e) { ar[e] = 0.f; an[e] = 0.f; }

  const float* xr = x + (size_t)n * CC;
#pragma unroll
  for (int k = 0; k < CC / 32; ++k) {
    int c = lane + 32 * k;
    float xv = xr[c];
    const float* wr = w_route + c * EE;
    const float* wn = w_noise + c * EE;
#pragma unroll
    for (int e = 0; e < EE; ++e) {
      ar[e] += xv * wr[e];
      an[e] += xv * wn[e];
    }
  }
#pragma unroll
  for (int off = 16; off >= 1; off >>= 1) {
#pragma unroll
    for (int e = 0; e < EE; ++e) {
      ar[e] += __shfl_xor(ar[e], off, 32);
      an[e] += __shfl_xor(an[e], off, 32);
    }
  }
  if (lane == 0) {
    float noisy[EE];
#pragma unroll
    for (int e = 0; e < EE; ++e) {
      float z = an[e] + b_noise[e];
      float sp = (z > 20.f) ? z : logf(1.f + expf(z));
      noisy[e] = ar[e] + b_route[e] + noise[(size_t)n * EE + e] * sp;
    }
    int i1 = 0; float v1 = noisy[0];
#pragma unroll
    for (int e = 1; e < EE; ++e)
      if (noisy[e] > v1) { v1 = noisy[e]; i1 = e; }
    float v2 = -3.4e38f;
#pragma unroll
    for (int e = 0; e < EE; ++e)
      if (e != i1 && noisy[e] > v2) v2 = noisy[e];
    top1[n] = i1;
    gate[n] = 1.f / (1.f + expf(v2 - v1));
  }
}

// ---------------------------------------------------------------------------
// Kernel 2: per-chunk expert histograms
// ---------------------------------------------------------------------------
__global__ __launch_bounds__(256) void k_hist(const int* __restrict__ top1,
                                              int* __restrict__ chunk_hist) {
  __shared__ int h[EE];
  int tid = threadIdx.x;
  if (tid < EE) h[tid] = 0;
  __syncthreads();
  int base = blockIdx.x * CHUNK;
  for (int i = tid; i < CHUNK; i += 256)
    atomicAdd(&h[top1[base + i]], 1);
  __syncthreads();
  if (tid < EE) chunk_hist[blockIdx.x * EE + tid] = h[tid];
}

// ---------------------------------------------------------------------------
// Kernel 3: serial exclusive scan over chunks per expert (tiny)
// ---------------------------------------------------------------------------
__global__ void k_scan(const int* __restrict__ chunk_hist,
                       int* __restrict__ chunk_base, int* __restrict__ nvalid) {
  int e = threadIdx.x;
  if (e >= EE) return;
  int run = 0;
  for (int c = 0; c < NCHUNK; ++c) {
    chunk_base[c * EE + e] = run;
    run += chunk_hist[c * EE + e];
  }
  nvalid[e] = run < CAP ? run : CAP;
}

// ---------------------------------------------------------------------------
// Kernel 4: slot assignment in global token order
// ---------------------------------------------------------------------------
__global__ __launch_bounds__(256) void k_assign(
    const int* __restrict__ top1, const float* __restrict__ gate,
    const int* __restrict__ chunk_base,
    int* __restrict__ idx_buf, float* __restrict__ gate_slot) {
  __shared__ int cnt[256][EE];
  int tid = threadIdx.x;
  int base = blockIdx.x * CHUNK;
  int lc[EE];
#pragma unroll
  for (int e = 0; e < EE; ++e) lc[e] = 0;
#pragma unroll
  for (int j = 0; j < 4; ++j) lc[top1[base + tid * 4 + j]]++;
#pragma unroll
  for (int e = 0; e < EE; ++e) cnt[tid][e] = lc[e];
  __syncthreads();
  if (tid < EE) {
    int run = 0;
    for (int t = 0; t < 256; ++t) {
      int v = cnt[t][tid];
      cnt[t][tid] = run;
      run += v;
    }
  }
  __syncthreads();
#pragma unroll
  for (int e = 0; e < EE; ++e) lc[e] = 0;
#pragma unroll
  for (int j = 0; j < 4; ++j) {
    int n = base + tid * 4 + j;
    int e = top1[n];
    int pos = chunk_base[blockIdx.x * EE + e] + cnt[tid][e] + lc[e];
    lc[e]++;
    if (pos < CAP) {
      idx_buf[e * CAP + pos] = n;
      gate_slot[e * CAP + pos] = gate[n];
    }
  }
}

// ---------------------------------------------------------------------------
// Kernel 5: weight transpose + fp32->bf16
// ---------------------------------------------------------------------------
__global__ __launch_bounds__(256) void k_conv_w1(const float* __restrict__ w1,
                                                 unsigned short* __restrict__ w1t) {
  int i = blockIdx.x * 256 + threadIdx.x;
  if (i >= EE * CC * DFF) return;
  int d = i % DFF;
  int c = (i / DFF) % CC;
  int e = i / (DFF * CC);
  w1t[((size_t)e * DFF + d) * CC + c] = f2bf(w1[i]);
}
__global__ __launch_bounds__(256) void k_conv_w2(const float* __restrict__ w2,
                                                 unsigned short* __restrict__ w2t) {
  int i = blockIdx.x * 256 + threadIdx.x;
  if (i >= EE * DFF * CC) return;
  int c = i % CC;
  int d = (i / CC) % DFF;
  int e = i / (DFF * CC);
  w2t[((size_t)e * CC + c) * DFF + d] = f2bf(w2[i]);
}

// ---------------------------------------------------------------------------
// Kernel 6: fused FFN with async-LDS weight staging + WMMA bf16.
// Per 32-wide D_FF chunk j:
//   [wait async0 + bar]  (w1 slab for j ready)
//   issue async stage of w2 slab(j)            } overlaps gemm1
//   gemm1 (fragment-double-buffered)           }
//   [bar]  issue async stage of w1 slab(j+32)  } overlaps ht store / gemm2
//   ht = bf16(relu(H+b1)^2)
//   [wait async6 (w2 done) + bar]
//   gemm2 (B-fragment double-buffered)
// ---------------------------------------------------------------------------
__global__ __launch_bounds__(256) void k_ffn(
    const float* __restrict__ x, const float* __restrict__ b1,
    const float* __restrict__ b2, const int* __restrict__ idx_buf,
    const float* __restrict__ gate_slot, const int* __restrict__ nvalid,
    const unsigned short* __restrict__ w1t, const unsigned short* __restrict__ w2t,
    float* __restrict__ out) {
  int e = blockIdx.x >> 7;
  int tile = blockIdx.x & 127;
  int rowbase = tile * MT;
  int nv = nvalid[e];
  if (rowbase >= nv) return;        // uniform across block

  __shared__ __align__(16) unsigned short xt[MT * XP];    // 64x384 bf16 x-tile (50 KB)
  __shared__ __align__(16) unsigned short ht[MT * HP];    // 64x32 bf16 H       ( 5 KB)
  __shared__ __align__(16) unsigned short w1s[32 * W1P];  // w1 slab 32x384     (24.5 KB)
  __shared__ __align__(16) unsigned short w2s[CC * HP];   // w2 slab 384x32     (30 KB)
  __shared__ float gs[MT];
  __shared__ int   tk[MT];

  int tid = threadIdx.x;
  const unsigned short* w1e = w1t + (size_t)e * DFF * CC;  // [d][c]
  const unsigned short* w2e = w2t + (size_t)e * CC * DFF;  // [c][d]
  const float* b1e = b1 + e * DFF;
  const float* b2e = b2 + e * CC;

  // prologue: kick off async stage of w1 slab for j=0 (overlaps x gather)
  {
#pragma unroll
    for (int s = 0; s < 6; ++s) {
      int g = tid + s * 256;          // 1536 granules of 16B: 32 rows x 48
      int d = g / 48, q = g - d * 48;
      async_cp16(w1e + (size_t)d * CC + q * 8, &w1s[d * W1P + q * 8]);
    }
  }

  if (tid < MT) {
    int slot = rowbase + tid;
    bool valid = slot < nv;
    tk[tid] = valid ? idx_buf[e * CAP + slot] : 0;
    gs[tid] = valid ? gate_slot[e * CAP + slot] : 0.f;
  }
  __syncthreads();

  // gather + convert x rows into LDS
  for (int i = tid; i < MT * 96; i += 256) {
    int r = i / 96, q = i % 96;
    uint2 pk;
    if (rowbase + r < nv) {
      float4 v = ((const float4*)(x + (size_t)tk[r] * CC))[q];
      pk.x = (unsigned int)f2bf(v.x) | ((unsigned int)f2bf(v.y) << 16);
      pk.y = (unsigned int)f2bf(v.z) | ((unsigned int)f2bf(v.w) << 16);
    } else {
      pk.x = 0u; pk.y = 0u;
    }
    *(uint2*)&xt[r * XP + q * 4] = pk;
  }

  int w    = tid >> 5;
  int lane = tid & 31;
  int ln   = lane & 15;
  int kh   = lane >> 4;
  int rt1  = w & 3;          // gemm1: 4 row tiles x 2 j-col tiles
  int ct1  = w >> 2;
  int rt2  = w & 3;          // gemm2: 4 row tiles x 2 column halves
  int cs2  = w >> 2;

  v8f acc2[12];
#pragma unroll
  for (int t = 0; t < 12; ++t) acc2[t] = (v8f){0.f,0.f,0.f,0.f,0.f,0.f,0.f,0.f};

  for (int j = 0; j < DFF; j += 32) {
    wait_async0();            // this wave's w1 slab portion complete
    __syncthreads();          // everyone's portions complete (+ x gather on j==0)

    // stage w2 slab for this j (consumed in gemm2 after the next two barriers)
#pragma unroll
    for (int s = 0; s < 6; ++s) {
      int g = tid + s * 256;          // 1536 granules: 384 rows x 4
      int r = g >> 2, q = g & 3;
      async_cp16(w2e + (size_t)r * DFF + j + q * 8, &w2s[r * HP + q * 8]);
    }

    // ---- gemm1: H(64x32) = x_tile @ w1s   (A/B fragments double-buffered)
    v8f acc1 = (v8f){0.f,0.f,0.f,0.f,0.f,0.f,0.f,0.f};
    int mA = rt1 * 16 + ln;
    int dloc = ct1 * 16 + ln;                 // local d within slab
    V16U A1[2], B1[2];
    A1[0].q[0] = *(const uint4*)&xt[mA * XP + 16 * kh];
    A1[0].q[1] = *(const uint4*)&xt[mA * XP + 16 * kh + 8];
    B1[0].q[0] = *(const uint4*)&w1s[dloc * W1P + 16 * kh];
    B1[0].q[1] = *(const uint4*)&w1s[dloc * W1P + 16 * kh + 8];
#pragma unroll
    for (int cb = 0; cb < CC; cb += 32) {
      int cur = (cb >> 5) & 1, nxt = cur ^ 1;
      if (cb + 32 < CC) {
        A1[nxt].q[0] = *(const uint4*)&xt[mA * XP + cb + 32 + 16 * kh];
        A1[nxt].q[1] = *(const uint4*)&xt[mA * XP + cb + 40 + 16 * kh];
        B1[nxt].q[0] = *(const uint4*)&w1s[dloc * W1P + cb + 32 + 16 * kh];
        B1[nxt].q[1] = *(const uint4*)&w1s[dloc * W1P + cb + 40 + 16 * kh];
      }
      acc1 = __builtin_amdgcn_wmma_f32_16x16x32_bf16(false, A1[cur].v, false, B1[cur].v,
                                                     (short)0, acc1, false, false);
    }
    float bias1 = b1e[j + ct1 * 16 + ln];
    __syncthreads();          // all gemm1 reads of w1s done; prev gemm2 done reading ht

    // stage w1 slab for j+32 (overwrites w1s; overlaps ht store + gemm2)
    if (j + 32 < DFF) {
#pragma unroll
      for (int s = 0; s < 6; ++s) {
        int g = tid + s * 256;
        int d = g / 48, q = g - d * 48;
        async_cp16(w1e + (size_t)(j + 32 + d) * CC + q * 8, &w1s[d * W1P + q * 8]);
      }
    }

#pragma unroll
    for (int v = 0; v < 8; ++v) {
      int m = rt1 * 16 + 8 * kh + v;
      float hv = acc1[v] + bias1;
      hv = fmaxf(hv, 0.f);
      hv = hv * hv;
      ht[m * HP + ct1 * 16 + ln] = f2bf(hv);
    }
    // w2 slab must be resident (<=6 outstanding are the just-issued w1 loads)
    if (j + 32 < DFF) wait_async6(); else wait_async0();
    __syncthreads();

    // ---- gemm2: acc2 += ht @ w2s   (B fragments double-buffered)
    V16U A2;
    int mA2 = rt2 * 16 + ln;
    A2.q[0] = *(const uint4*)&ht[mA2 * HP + 16 * kh];
    A2.q[1] = *(const uint4*)&ht[mA2 * HP + 16 * kh + 8];
    int nc0 = (cs2 * 12) * 16 + ln;
    V16U B2[2];
    B2[0].q[0] = *(const uint4*)&w2s[nc0 * HP + 16 * kh];
    B2[0].q[1] = *(const uint4*)&w2s[nc0 * HP + 16 * kh + 8];
#pragma unroll
    for (int t = 0; t < 12; ++t) {
      int cur = t & 1, nxt = cur ^ 1;
      if (t + 1 < 12) {
        int ncol = (cs2 * 12 + t + 1) * 16 + ln;
        B2[nxt].q[0] = *(const uint4*)&w2s[ncol * HP + 16 * kh];
        B2[nxt].q[1] = *(const uint4*)&w2s[ncol * HP + 16 * kh + 8];
      }
      acc2[t] = __builtin_amdgcn_wmma_f32_16x16x32_bf16(false, A2.v, false, B2[cur].v,
                                                        (short)0, acc2[t], false, false);
    }
  }

  // ---- epilogue: +b2, gate, scatter rows to out
#pragma unroll
  for (int t = 0; t < 12; ++t) {
    int ncol = (cs2 * 12 + t) * 16 + ln;
    float bias2 = b2e[ncol];
#pragma unroll
    for (int v = 0; v < 8; ++v) {
      int m = rt2 * 16 + 8 * kh + v;
      if (rowbase + m < nv) {
        float y = (acc2[t][v] + bias2) * gs[m];
        out[(size_t)tk[m] * CC + ncol] = y;
      }
    }
  }
}

// ---------------------------------------------------------------------------
// Host launcher
// ---------------------------------------------------------------------------
extern "C" void kernel_launch(void* const* d_in, const int* in_sizes, int n_in,
                              void* d_out, int out_size, void* d_ws, size_t ws_size,
                              hipStream_t stream) {
  const float* x       = (const float*)d_in[0];
  const float* noise   = (const float*)d_in[1];
  const float* w_route = (const float*)d_in[2];
  const float* b_route = (const float*)d_in[3];
  const float* w_noise = (const float*)d_in[4];
  const float* b_noise = (const float*)d_in[5];
  const float* w1      = (const float*)d_in[6];
  const float* b1      = (const float*)d_in[7];
  const float* w2      = (const float*)d_in[8];
  const float* b2      = (const float*)d_in[9];
  float* out = (float*)d_out;

  char* ws = (char*)d_ws;
  size_t off = 0;
  auto alloc = [&](size_t bytes) {
    char* p = ws + off;
    off = (off + bytes + 255) & ~(size_t)255;
    return p;
  };
  int*   top1       = (int*)  alloc(NN * sizeof(int));
  float* gate       = (float*)alloc(NN * sizeof(float));
  int*   chunk_hist = (int*)  alloc(NCHUNK * EE * sizeof(int));
  int*   chunk_base = (int*)  alloc(NCHUNK * EE * sizeof(int));
  int*   nvalid     = (int*)  alloc(EE * sizeof(int));
  int*   idx_buf    = (int*)  alloc((size_t)EE * CAP * sizeof(int));
  float* gate_slot  = (float*)alloc((size_t)EE * CAP * sizeof(float));
  unsigned short* w1t = (unsigned short*)alloc((size_t)EE * CC * DFF * 2);
  unsigned short* w2t = (unsigned short*)alloc((size_t)EE * CC * DFF * 2);

  int n4 = out_size / 4;
  k_zero<<<(n4 + 255) / 256, 256, 0, stream>>>(out, n4);
  int nw = EE * CC * DFF;
  k_conv_w1<<<(nw + 255) / 256, 256, 0, stream>>>(w1, w1t);
  k_conv_w2<<<(nw + 255) / 256, 256, 0, stream>>>(w2, w2t);
  k_route<<<NN / 8, 256, 0, stream>>>(x, noise, w_route, b_route, w_noise,
                                      b_noise, top1, gate);
  k_hist<<<NCHUNK, 256, 0, stream>>>(top1, chunk_hist);
  k_scan<<<1, 32, 0, stream>>>(chunk_hist, chunk_base, nvalid);
  k_assign<<<NCHUNK, 256, 0, stream>>>(top1, gate, chunk_base, idx_buf, gate_slot);
  k_ffn<<<EE * (CAP / MT), 256, 0, stream>>>(x, b1, b2, idx_buf, gate_slot,
                                             nvalid, w1t, w2t, out);
}
